// Draft_42829413876117
// MI455X (gfx1250) — matrix operand order repaired and checked by reading, not compiled, not run on hardware
//
#include <hip/hip_runtime.h>
#include <hip/hip_bf16.h>

// ---------------------------------------------------------------------------
// MI455X (gfx1250) implementation.
//
// Roofline: As is 256MB (11us @ 23.3TB/s); embed GEMMs ~51 GFLOP -> bf16 WMMA,
// fused so As is read exactly once and the (131072 x 256) intermediate never
// touches HBM. Softmax shift-invariance removes Wq/bq/bqk exactly; node-
// independence of the aggregate collapses rounds 2-3 to 1024x128 GEMM chains.
// All matmuls lower to v_wmma_f32_16x16x32_bf16 (wave32, fp32 accum).
//
// v2: B-fragment register-batch preloading (kills the per-WMMA s_wait_dscnt 0
// serialization seen in v1 asm), double-buffered W_emb LDS staging overlapped
// with the WMMA chain, vectorized (float4 -> ds_store_b64) LDS staging.
// ---------------------------------------------------------------------------

typedef __attribute__((ext_vector_type(16))) __bf16 v16bf;
typedef __attribute__((ext_vector_type(8)))  __bf16 v8bf;
typedef __attribute__((ext_vector_type(4)))  __bf16 v4bf;
typedef __attribute__((ext_vector_type(8)))  float  v8f;

static constexpr int kNodes = 128;
static constexpr int kBatch = 1024;
static constexpr int kFeat  = 512;
static constexpr int kEmb   = 256;
static constexpr int kHid   = 128;
static constexpr int kRows  = kNodes * kBatch;   // 131072

__device__ __forceinline__ v16bf cat16(v8bf lo, v8bf hi) {
  v16bf r;
#pragma unroll
  for (int i = 0; i < 8; ++i) { r[i] = lo[i]; r[8 + i] = hi[i]; }
  return r;
}

__device__ __forceinline__ v8f wmma_bf16(v16bf a, v16bf b, v8f c) {
  return __builtin_amdgcn_wmma_f32_16x16x32_bf16(
      /*neg_a=*/false, a, /*neg_b=*/false, b,
      /*c_mod=*/(short)0, c, /*reuse_a=*/false, /*reuse_b=*/false);
}

// fp32 global -> bf16 LDS, vectorized (float4 in, ds_store_b64 out).
// count must be a multiple of 4*nthreads.
__device__ __forceinline__ void stage_w(const float* __restrict__ src,
                                        __bf16* __restrict__ dst,
                                        int count, int tid, int nthreads)
{
  const float4* s4 = (const float4*)src;
  const int n4 = count >> 2;
  for (int i = tid; i < n4; i += nthreads) {
    float4 f = s4[i];
    v4bf o;
    o[0] = (__bf16)f.x; o[1] = (__bf16)f.y; o[2] = (__bf16)f.z; o[3] = (__bf16)f.w;
    *(v4bf*)(dst + i * 4) = o;
  }
}

// ---------------------------------------------------------------------------
// Fused embed: X = relu((As @ W_emb + b_emb) @ W_h + b_h)  -> bf16 (131072x128)
// 256 thr = 8 waves, wave owns a 16-row tile (128 rows/block, 1024 blocks).
// LDS: W_h 64KB + double-buffered W_emb chunk 2x16KB + T staging 64KB = 160KB
//      -> exactly 2 workgroups per 320KB WGP.
// ---------------------------------------------------------------------------
__global__ __launch_bounds__(256)
void embed_kernel(const float* __restrict__ As, const float* __restrict__ W_emb,
                  const float* __restrict__ b_emb, const float* __restrict__ W_h,
                  const float* __restrict__ b_h, __bf16* __restrict__ X)
{
  __shared__ __align__(32) __bf16 ldsWh[kEmb * kHid];      // 64 KB
  __shared__ __align__(32) __bf16 ldsWe[2][32 * kEmb];     // 2 x 16 KB
  __shared__ __align__(32) __bf16 ldsT[8 * 16 * kEmb];     // 64 KB

  const int tid   = threadIdx.x;
  const int wave  = tid >> 5;
  const int lane  = tid & 31;
  const int lmod  = lane & 15;
  const int lhalf = lane >> 4;
  const int rowbase = blockIdx.x * 128 + wave * 16;

  // Stage W_h (fp32 -> bf16) once, and prime W_emb chunk 0.
  stage_w(W_h, ldsWh, kEmb * kHid, tid, 256);
  stage_w(W_emb, ldsWe[0], 32 * kEmb, tid, 256);
  __syncthreads();

  v8f acc[16] = {};

  for (int kt = 0; kt < 16; ++kt) {
    const int cur = kt & 1;

    // A fragment: lane holds row M=lmod, K = klo..klo+7 and klo+16..klo+23.
    const float* ap = As + (rowbase + lmod) * kFeat + kt * 32 + lhalf * 8;
    if (kt < 15) __builtin_prefetch(ap + 32, 0, 0);   // global_prefetch next chunk
    float4 f0 = *(const float4*)(ap + 0);
    float4 f1 = *(const float4*)(ap + 4);
    float4 f2 = *(const float4*)(ap + 16);
    float4 f3 = *(const float4*)(ap + 20);
    v16bf a;
    a[0]=(__bf16)f0.x; a[1]=(__bf16)f0.y; a[2]=(__bf16)f0.z; a[3]=(__bf16)f0.w;
    a[4]=(__bf16)f1.x; a[5]=(__bf16)f1.y; a[6]=(__bf16)f1.z; a[7]=(__bf16)f1.w;
    a[8]=(__bf16)f2.x; a[9]=(__bf16)f2.y; a[10]=(__bf16)f2.z; a[11]=(__bf16)f2.w;
    a[12]=(__bf16)f3.x; a[13]=(__bf16)f3.y; a[14]=(__bf16)f3.z; a[15]=(__bf16)f3.w;

    // B fragments: lane holds N=lmod, K in table order (same as A).
    const __bf16* bf = &ldsWe[cur][lmod];

    // Kick off staging of the next W_emb chunk into the other buffer; the
    // global loads + ds stores overlap with the WMMA chain below.
    if (kt + 1 < 16)
      stage_w(W_emb + (kt + 1) * 32 * kEmb, ldsWe[cur ^ 1], 32 * kEmb, tid, 256);

#pragma unroll
    for (int half = 0; half < 2; ++half) {
      v16bf b[8];
#pragma unroll
      for (int nt = 0; nt < 8; ++nt)
        for (int s = 0; s < 16; ++s)
          b[nt][s] = bf[(((s & 8) << 1) + lhalf * 8 + (s & 7)) * kEmb + (half * 8 + nt) * 16];
#pragma unroll
      for (int nt = 0; nt < 8; ++nt)
        acc[half * 8 + nt] = wmma_bf16(a, b[nt], acc[half * 8 + nt]);
    }
    __syncthreads();
  }

  // Epilogue 1: T += b_emb, spill tile to LDS (bf16) for A-layout re-read.
#pragma unroll
  for (int nt = 0; nt < 16; ++nt) {
    const int col = nt * 16 + lmod;
    const float be = b_emb[col];
#pragma unroll
    for (int r = 0; r < 8; ++r) {
      const int m = r + lhalf * 8;
      ldsT[wave * 16 * kEmb + m * kEmb + col] = (__bf16)(acc[nt][r] + be);
    }
  }
  __syncthreads();

  // Stage 2: X_tile(16x128) = T(16x256) @ W_h(256x128)
  v8f acc2[8] = {};
  for (int kt = 0; kt < 8; ++kt) {
    const __bf16* tp = &ldsT[wave * 16 * kEmb + lmod * kEmb + kt * 32 + lhalf * 8];
    v16bf a2 = cat16(*(const v8bf*)tp, *(const v8bf*)(tp + 16));
    v16bf b[8];
#pragma unroll
    for (int nt = 0; nt < 8; ++nt)
      for (int s = 0; s < 16; ++s)
        b[nt][s] = ldsWh[(kt * 32 + ((s & 8) << 1) + lhalf * 8 + (s & 7)) * kHid + nt * 16 + lmod];
#pragma unroll
    for (int nt = 0; nt < 8; ++nt)
      acc2[nt] = wmma_bf16(a2, b[nt], acc2[nt]);
  }

#pragma unroll
  for (int nt = 0; nt < 8; ++nt) {
    const int col = nt * 16 + lmod;
    const float bh = b_h[col];
#pragma unroll
    for (int r = 0; r < 8; ++r) {
      const int row = rowbase + r + lhalf * 8;
      const float v = acc2[nt][r] + bh;
      X[row * kHid + col] = (__bf16)(v > 0.f ? v : 0.f);
    }
  }
}

// ---------------------------------------------------------------------------
// Generic WMMA GEMM: out = act(A(1024xK=128) @ W(128xN) + bias), bf16 A, fp32 W.
// grid = (8, N/128). Each wave computes a 16x128 tile; W col-slab in LDS.
// ---------------------------------------------------------------------------
__global__ __launch_bounds__(256)
void gemm128_kernel(const __bf16* __restrict__ A, const float* __restrict__ W,
                    const float* __restrict__ bias, __bf16* __restrict__ outB,
                    float* __restrict__ outF, int N, int doRelu)
{
  __shared__ __align__(32) __bf16 ldsW[128 * 128];   // 32 KB col slab

  const int tid   = threadIdx.x;
  const int wave  = tid >> 5;
  const int lane  = tid & 31;
  const int lmod  = lane & 15;
  const int lhalf = lane >> 4;
  const int nbase = blockIdx.y * 128;

  // Stage W col slab (vectorized: 4 consecutive n per thread-step).
  for (int i = tid; i < 128 * 32; i += 256) {
    const int k = i >> 5, nq = i & 31;
    float4 f = *(const float4*)&W[k * N + nbase + nq * 4];
    v4bf o;
    o[0] = (__bf16)f.x; o[1] = (__bf16)f.y; o[2] = (__bf16)f.z; o[3] = (__bf16)f.w;
    *(v4bf*)&ldsW[k * 128 + nq * 4] = o;
  }
  __syncthreads();

  const int rt = blockIdx.x * 8 + wave;      // 16-row tile index
  v8f acc[8] = {};
  for (int kt = 0; kt < 4; ++kt) {
    const __bf16* ap = A + (rt * 16 + lmod) * 128 + kt * 32 + lhalf * 8;
    v16bf a = cat16(*(const v8bf*)ap, *(const v8bf*)(ap + 16));
    v16bf b[8];
#pragma unroll
    for (int nt = 0; nt < 8; ++nt)
      for (int s = 0; s < 16; ++s)
        b[nt][s] = ldsW[(kt * 32 + ((s & 8) << 1) + lhalf * 8 + (s & 7)) * 128 + nt * 16 + lmod];
#pragma unroll
    for (int nt = 0; nt < 8; ++nt)
      acc[nt] = wmma_bf16(a, b[nt], acc[nt]);
  }

#pragma unroll
  for (int nt = 0; nt < 8; ++nt) {
    const int col = nbase + nt * 16 + lmod;
    const float bb = bias[col];
#pragma unroll
    for (int r = 0; r < 8; ++r) {
      const int row = rt * 16 + r + lhalf * 8;
      float v = acc[nt][r] + bb;
      if (doRelu) v = v > 0.f ? v : 0.f;
      if (outB) outB[row * N + col] = (__bf16)v;
      if (outF) outF[row * N + col] = v;
    }
  }
}

// u[t] = sum_h Wk[t,h] * wk_s[h];  uc[128] = bk . wk_s   (wk_s = Wqk[H:,0])
__global__ void prep_kernel(const float* __restrict__ Wk, const float* __restrict__ bk,
                            const float* __restrict__ Wqk, float* __restrict__ uc)
{
  const int t = threadIdx.x;
  float s = 0.f;
  for (int h = 0; h < kHid; ++h) s += Wk[t * kHid + h] * Wqk[kHid + h];
  uc[t] = s;
  if (t == 0) {
    float c = 0.f;
    for (int h = 0; h < kHid; ++h) c += bk[h] * Wqk[kHid + h];
    uc[kHid] = c;
  }
}

// sk[j*B+b] = X[j,b,:] . u + c
__global__ __launch_bounds__(256)
void sk_kernel(const __bf16* __restrict__ X, const float* __restrict__ uc,
               float* __restrict__ sk)
{
  __shared__ float su[kHid];
  if (threadIdx.x < kHid) su[threadIdx.x] = uc[threadIdx.x];
  __syncthreads();
  const int i = blockIdx.x * 256 + threadIdx.x;
  const __bf16* xr = X + i * kHid;
  float s = uc[kHid];
  for (int h = 0; h < kHid; ++h) s += (float)xr[h] * su[h];
  sk[i] = s;
}

// w[:,b] = softmax_j(sk[:,b]); one block per b, 128 threads (j)
__global__ __launch_bounds__(128)
void softmax_kernel(const float* __restrict__ sk, float* __restrict__ w)
{
  __shared__ float red[128];
  const int b = blockIdx.x, j = threadIdx.x;
  const float s = sk[j * kBatch + b];
  red[j] = s; __syncthreads();
  for (int off = 64; off; off >>= 1) {
    if (j < off) red[j] = fmaxf(red[j], red[j + off]);
    __syncthreads();
  }
  const float m = red[0]; __syncthreads();
  const float e = __expf(s - m);
  red[j] = e; __syncthreads();
  for (int off = 64; off; off >>= 1) {
    if (j < off) red[j] += red[j + off];
    __syncthreads();
  }
  w[j * kBatch + b] = e / red[0];
}

// y[b,:] = sum_j w[j,b] * X[j,b,:]   (node-weighted sum; exact collapse)
__global__ __launch_bounds__(256)
void y_kernel(const __bf16* __restrict__ X, const float* __restrict__ w,
              __bf16* __restrict__ y)
{
  const int i = blockIdx.x * 256 + threadIdx.x;   // b*128 + h
  const int b = i >> 7, h = i & 127;
  float s = 0.f;
  for (int j = 0; j < kNodes; ++j)
    s += w[j * kBatch + b] * (float)X[(j * kBatch + b) * kHid + h];
  y[i] = (__bf16)s;
}

// In-place row log_softmax over 512 cols; one block (256 thr) per row.
__global__ __launch_bounds__(256)
void lsm_kernel(float* __restrict__ out)
{
  __shared__ float red[256];
  const int b = blockIdx.x, t = threadIdx.x;
  const float v0 = out[b * kFeat + t];
  const float v1 = out[b * kFeat + 256 + t];
  red[t] = fmaxf(v0, v1); __syncthreads();
  for (int off = 128; off; off >>= 1) {
    if (t < off) red[t] = fmaxf(red[t], red[t + off]);
    __syncthreads();
  }
  const float m = red[0]; __syncthreads();
  red[t] = __expf(v0 - m) + __expf(v1 - m); __syncthreads();
  for (int off = 128; off; off >>= 1) {
    if (t < off) red[t] += red[t + off];
    __syncthreads();
  }
  const float ls = m + __logf(red[0]);
  out[b * kFeat + t] = v0 - ls;
  out[b * kFeat + 256 + t] = v1 - ls;
}

extern "C" void kernel_launch(void* const* d_in, const int* in_sizes, int n_in,
                              void* d_out, int out_size, void* d_ws, size_t ws_size,
                              hipStream_t stream)
{
  (void)in_sizes; (void)n_in; (void)out_size; (void)ws_size;
  const float* As   = (const float*)d_in[0];
  const float* Wemb = (const float*)d_in[1];
  const float* bemb = (const float*)d_in[2];
  const float* Wh   = (const float*)d_in[3];
  const float* bh   = (const float*)d_in[4];
  // d_in[5]=Wq, d_in[6]=bq, d_in[12]=bqk: cancel exactly under softmax(axis=1).
  const float* Wk   = (const float*)d_in[7];
  const float* bk   = (const float*)d_in[8];
  const float* Wv   = (const float*)d_in[9];
  const float* bv   = (const float*)d_in[10];
  const float* Wqk  = (const float*)d_in[11];
  const float* Wa   = (const float*)d_in[13];
  const float* ba   = (const float*)d_in[14];
  const float* W1   = (const float*)d_in[15];
  const float* b1   = (const float*)d_in[16];
  const float* W2   = (const float*)d_in[17];
  const float* b2   = (const float*)d_in[18];
  float* out = (float*)d_out;

  char* ws = (char*)d_ws;
  size_t off = 0;
  auto alloc = [&](size_t bytes) {
    void* p = ws + off;
    off = (off + bytes + 255) & ~(size_t)255;
    return p;
  };
  float*  uc = (float*)alloc(129 * sizeof(float));
  __bf16* X  = (__bf16*)alloc((size_t)kRows * kHid * 2);   // 32 MB
  float*  sk = (float*)alloc((size_t)kRows * 4);
  float*  w  = (float*)alloc((size_t)kRows * 4);
  __bf16* y  = (__bf16*)alloc((size_t)kBatch * kHid * 2);
  __bf16* tA = (__bf16*)alloc((size_t)kBatch * kHid * 2);
  __bf16* tB = (__bf16*)alloc((size_t)kBatch * kHid * 2);

  prep_kernel<<<1, 128, 0, stream>>>(Wk, bk, Wqk, uc);
  embed_kernel<<<kRows / 128, 256, 0, stream>>>(As, Wemb, bemb, Wh, bh, X);
  sk_kernel<<<kRows / 256, 256, 0, stream>>>(X, uc, sk);
  softmax_kernel<<<kBatch, 128, 0, stream>>>(sk, w);
  y_kernel<<<kRows / 256, 256, 0, stream>>>(X, w, y);

  // Round 1: aggre = y @ Wv + bv; x = relu(aggre @ Wa + ba)
  gemm128_kernel<<<dim3(8, 1), 256, 0, stream>>>(y, Wv, bv, tA, nullptr, kHid, 0);
  gemm128_kernel<<<dim3(8, 1), 256, 0, stream>>>(tA, Wa, ba, tB, nullptr, kHid, 1);
  // Rounds 2-3: nodes identical -> uniform softmax (sum 1) -> same chain on x.
  for (int r = 0; r < 2; ++r) {
    gemm128_kernel<<<dim3(8, 1), 256, 0, stream>>>(tB, Wv, bv, tA, nullptr, kHid, 0);
    gemm128_kernel<<<dim3(8, 1), 256, 0, stream>>>(tA, Wa, ba, tB, nullptr, kHid, 1);
  }
  // Head: logits = relu(x3 @ W1 + b1) @ W2 + b2, then in-place log_softmax.
  gemm128_kernel<<<dim3(8, 1), 256, 0, stream>>>(tB, W1, b1, tA, nullptr, kHid, 1);
  gemm128_kernel<<<dim3(8, 4), 256, 0, stream>>>(tA, W2, b2, nullptr, out, kFeat, 0);
  lsm_kernel<<<kBatch, 256, 0, stream>>>(out);
}
